// PatchySan_37890201485650
// MI455X (gfx1250) — compile-verified, matrix operand order checked by reading, not run on hardware
//
#include <hip/hip_runtime.h>
#include <hip/hip_bf16.h>

typedef __bf16 bf16;
typedef __attribute__((ext_vector_type(16))) __bf16 v16bf;
typedef __attribute__((ext_vector_type(2)))  __bf16 bf16x2;
typedef __attribute__((ext_vector_type(8)))  float   v8f;
typedef __attribute__((ext_vector_type(4)))  unsigned int u32x4;
typedef __attribute__((ext_vector_type(8)))  int     i32x8;
typedef __attribute__((ext_vector_type(4)))  int     i32x4;

#define WMMA_BF16(a, b, c) \
  __builtin_amdgcn_wmma_f32_16x16x32_bf16(false, (a), false, (b), (short)0, (c), false, false)

// Problem constants (match reference)
constexpr int BG   = 256;            // graphs
constexpr int WNUM = 64;             // frames per graph
constexpr int KNB  = 16;             // neighbours per frame
constexpr int DIN  = 128;
constexpr int DD   = 256;
constexpr int HH   = 1024;
constexpr int NNODE = BG * 128;      // 32768
constexpr int FR   = BG * WNUM;      // 16384 frame rows
constexpr int KC   = KNB * DD;       // 4096
constexpr int KTOT = WNUM * DD;      // 16384

// LDS tile geometry for TDM-staged B operands:
// 64 rows x 128 K bf16 = 256B/row, padded +16B/row by the TDM -> 272B row pitch
constexpr int KSPAN   = 128;
constexpr int ROWPAD  = 136;         // bf16 elements per padded row (272 B)

// ---------------------------------------------------------------------------
// Tensor Data Mover: 2D tile load Global->LDS (D# per ISA 8.3/8.4).
// data_size=2B; pad_enable, pad_interval=64 dwords (256B), pad_amount=4 dwords.
// ---------------------------------------------------------------------------
__device__ __forceinline__ unsigned lds_offset_of(const void* p) {
  // Generic LDS-aperture address: low 32 bits are the LDS byte offset.
  return (unsigned)(unsigned long long)p;
}

__device__ __forceinline__ void tdm_load_2d(unsigned lds_off, const void* gaddr,
                                            unsigned td0, unsigned td1,
                                            unsigned tile0, unsigned tile1,
                                            unsigned stride0) {
  unsigned long long ga = (unsigned long long)gaddr;
  u32x4 g0;
  g0[0] = 1u;                                   // count=1, user descriptor
  g0[1] = lds_off;                              // lds_addr (bytes)
  g0[2] = (unsigned)ga;                         // global_addr[31:0]
  g0[3] = (unsigned)(ga >> 32) | 0x80000000u;   // global_addr[56:32] | type=2
  i32x8 g1;
  g1[0] = (1 << 16) | (1 << 20) | (5 << 22) | (3 << 25); // 2B, pad 4dw/64dw
  g1[1] = (int)((td0 & 0xFFFFu) << 16);                  // tensor_dim0 lo
  g1[2] = (int)((td0 >> 16) | ((td1 & 0xFFFFu) << 16));  // td0 hi | td1 lo
  g1[3] = (int)((td1 >> 16) | (tile0 << 16));            // td1 hi | tile_dim0
  g1[4] = (int)tile1;                                    // tile_dim1 (tile2=0)
  g1[5] = (int)stride0;                                  // dim0 stride lo
  g1[6] = 0;                                             // stride hi / s1 lo
  g1[7] = 0;
  i32x4 z4 = {0, 0, 0, 0};
#if __clang_major__ >= 23
  i32x8 z8 = {};
  __builtin_amdgcn_tensor_load_to_lds(g0, g1, z4, z4, z8, 0);
#else
  __builtin_amdgcn_tensor_load_to_lds(g0, g1, z4, z4, 0);
#endif
}

// ---------------------------------------------------------------------------
// Fragment helpers. 16-bit A-matrix 16x32 layout (ISA 7.12.2):
//   lane = half*16 + r ; r indexes the row (A) / column (B-transposed),
//   half picks K sub-block {0..7} vs {8..15}; VGPRs 4..7 hold K+16.
// p must point at element (row, k0 + half*8).
// ---------------------------------------------------------------------------
__device__ __forceinline__ v16bf frag_from_bf16(const bf16* p) {
  v16bf f;
#pragma unroll
  for (int j = 0; j < 4; ++j) {
    bf16x2 lo = *(const bf16x2*)(p + 2 * j);
    bf16x2 hi = *(const bf16x2*)(p + 16 + 2 * j);
    f[2 * j]     = lo.x; f[2 * j + 1]     = lo.y;
    f[8 + 2 * j] = hi.x; f[8 + 2 * j + 1] = hi.y;
  }
  return f;
}

__device__ __forceinline__ v16bf frag_from_f32(const float* p) {
  v16bf f;
#pragma unroll
  for (int j = 0; j < 4; ++j) {
    f[2 * j]         = (bf16)p[2 * j];
    f[2 * j + 1]     = (bf16)p[2 * j + 1];
    f[8 + 2 * j]     = (bf16)p[16 + 2 * j];
    f[8 + 2 * j + 1] = (bf16)p[16 + 2 * j + 1];
  }
  return f;
}

__device__ __forceinline__ v16bf frag_zero() {
  v16bf f;
#pragma unroll
  for (int e = 0; e < 16; ++e) f[e] = (bf16)0.0f;
  return f;
}

// D layout: VGPR v -> M = v + 8*half, N = lane&15 (ISA 7.12.2 table)
__device__ __forceinline__ void store_tile_bf16(bf16* C, int ldc, int m0, int n0,
                                                const v8f& acc, const float* bias,
                                                bool relu_en, int lane) {
  int half = lane >> 4, n = n0 + (lane & 15);
  float bv = bias[n];
#pragma unroll
  for (int v = 0; v < 8; ++v) {
    float val = acc[v] + bv;
    if (relu_en) val = fmaxf(val, 0.0f);
    C[(long)(m0 + v + 8 * half) * ldc + n] = (bf16)val;
  }
}

__device__ __forceinline__ void store_tile_f32(float* C, int ldc, int m0, int n0,
                                               const v8f& acc, int lane) {
  int half = lane >> 4, n = n0 + (lane & 15);
#pragma unroll
  for (int v = 0; v < 8; ++v)
    C[(long)(m0 + v + 8 * half) * ldc + n] = acc[v];
}

// ---------------------------------------------------------------------------
// Weight prep: W[K][N] f32 -> Wt[N][K] bf16 (so B fragments use the A layout)
// ---------------------------------------------------------------------------
__global__ void k_transpose_bf16(const float* __restrict__ src, bf16* __restrict__ dst,
                                 int K, int NN) {
  int t = blockIdx.x * blockDim.x + threadIdx.x;
  if (t < K * NN) {
    int n = t / K;
    int k = t - n * K;
    dst[t] = (bf16)src[(long)k * NN + n];
  }
}

// ---------------------------------------------------------------------------
// Stage 1: x[32768][256] = bf16(node_attr @ W_node + b_node)
// ---------------------------------------------------------------------------
__global__ __launch_bounds__(128) void k_embed(const float* __restrict__ na,
                                               const bf16* __restrict__ WnT,
                                               const float* __restrict__ bn,
                                               bf16* __restrict__ x) {
  int lane = threadIdx.x & 31;
  int gw = blockIdx.x * (blockDim.x >> 5) + (threadIdx.x >> 5);
  int mt = gw >> 2, ns = gw & 3;                     // 2048 m-tiles, 4 n-strips
  int m0 = mt * 16, n0 = ns * 64;
  int half = lane >> 4, r = lane & 15;
  v8f acc[4] = {};
#pragma unroll
  for (int k0 = 0; k0 < DIN; k0 += 32) {
    v16bf a = frag_from_f32(na + (long)(m0 + r) * DIN + k0 + half * 8);
#pragma unroll
    for (int t = 0; t < 4; ++t) {
      v16bf b = frag_from_bf16(WnT + (long)(n0 + t * 16 + r) * DIN + k0 + half * 8);
      acc[t] = WMMA_BF16(a, b, acc[t]);
    }
  }
#pragma unroll
  for (int t = 0; t < 4; ++t)
    store_tile_bf16(x, DD, m0, n0 + t * 16, acc[t], bn, false, lane);
}

// ---------------------------------------------------------------------------
// Stage 2: hc[16384][256] = relu(gather(frame_idx, x) @ W_conv + b_conv)
// Block = 4 waves: same 64-wide n-strip, 4 consecutive m-tiles.
// B (W_conv^T) tiles are TDM-streamed into double-buffered LDS; the fused
// gather selects one neighbour node per 256-wide K slot; masked rows -> 0.
// ---------------------------------------------------------------------------
__global__ __launch_bounds__(128) void k_conv(const bf16* __restrict__ x,
                                              const int* __restrict__ fidx,
                                              const bf16* __restrict__ WcT,
                                              const float* __restrict__ bc,
                                              bf16* __restrict__ hc) {
  __shared__ __align__(16) bf16 smem[2][64 * ROWPAD];
  int lane = threadIdx.x & 31;
  int wave = threadIdx.x >> 5;
  int nb = blockIdx.x & 3;                           // n-strip (4)
  int mg = blockIdx.x >> 2;                          // m-group (256)
  int mt = mg * 4 + wave;
  int m0 = mt * 16, n0 = nb * 64;
  int half = lane >> 4, r = lane & 15;
  const int* frow = fidx + (long)(m0 + r) * KNB;
  const bf16* gB = WcT + (long)n0 * KC;              // B rows for this strip
  bool w0 = (wave == 0);
  constexpr int NSPAN = KC / KSPAN;                  // 32 spans of 128 K

  if (w0) tdm_load_2d(lds_offset_of(&smem[0][0]), gB, KC, DD, KSPAN, 64, KC);

  v8f acc[4] = {};
  for (int s = 0; s < NSPAN; ++s) {
    if (w0) {
      if (s + 1 < NSPAN) {
        tdm_load_2d(lds_offset_of(&smem[(s + 1) & 1][0]), gB + (s + 1) * KSPAN,
                    KC, DD, KSPAN, 64, KC);
        __builtin_amdgcn_s_wait_tensorcnt(1);
      } else {
        __builtin_amdgcn_s_wait_tensorcnt(0);
      }
    }
    __syncthreads();
    const bf16* bbuf = smem[s & 1];
    int node = frow[s >> 1];                         // 256-wide neighbour slot
    bool valid = node >= 0;
    const bf16* xr = x + (long)(valid ? node : 0) * DD + (s & 1) * KSPAN;
#pragma unroll
    for (int c = 0; c < 4; ++c) {                    // 4 k-chunks of 32
      v16bf a = valid ? frag_from_bf16(xr + c * 32 + half * 8) : frag_zero();
#pragma unroll
      for (int t = 0; t < 4; ++t) {
        v16bf b = frag_from_bf16(bbuf + (t * 16 + r) * ROWPAD + c * 32 + half * 8);
        acc[t] = WMMA_BF16(a, b, acc[t]);
      }
    }
    __syncthreads();
  }
#pragma unroll
  for (int t = 0; t < 4; ++t)
    store_tile_bf16(hc, DD, m0, n0 + t * 16, acc[t], bc, true, lane);
}

// ---------------------------------------------------------------------------
// Stage 3 (split-K): part[ks][256][1024] = A[256][16384] @ W_h0
// A is the contiguous reinterpretation of hc (reshape is free).
// Same TDM-staged LDS double buffering for B; block = 4 m-tiles x 1 n-strip.
// ---------------------------------------------------------------------------
__global__ __launch_bounds__(128) void k_h0_part(const bf16* __restrict__ A,
                                                 const bf16* __restrict__ W,
                                                 float* __restrict__ part) {
  __shared__ __align__(16) bf16 smem[2][64 * ROWPAD];
  int lane = threadIdx.x & 31;
  int wave = threadIdx.x >> 5;
  int ks = blockIdx.x & 7;                           // 8-way split-K
  int ns = (blockIdx.x >> 3) & 15;                   // 16 n-strips
  int mg = blockIdx.x >> 7;                          // 4 m-groups
  int mt = mg * 4 + wave;                            // 16 m-tiles
  int m0 = mt * 16, n0 = ns * 64;
  int half = lane >> 4, r = lane & 15;
  int kbeg = ks * (KTOT / 8);
  const bf16* gB = W + (long)n0 * KTOT + kbeg;
  bool w0 = (wave == 0);
  constexpr int NSPAN = (KTOT / 8) / KSPAN;          // 16 spans

  if (w0) tdm_load_2d(lds_offset_of(&smem[0][0]), gB, KTOT, HH, KSPAN, 64, KTOT);

  v8f acc[4] = {};
  for (int s = 0; s < NSPAN; ++s) {
    if (w0) {
      if (s + 1 < NSPAN) {
        tdm_load_2d(lds_offset_of(&smem[(s + 1) & 1][0]), gB + (s + 1) * KSPAN,
                    KTOT, HH, KSPAN, 64, KTOT);
        __builtin_amdgcn_s_wait_tensorcnt(1);
      } else {
        __builtin_amdgcn_s_wait_tensorcnt(0);
      }
    }
    __syncthreads();
    const bf16* bbuf = smem[s & 1];
    const bf16* ar = A + (long)(m0 + r) * KTOT + kbeg + s * KSPAN;
#pragma unroll
    for (int c = 0; c < 4; ++c) {
      v16bf a = frag_from_bf16(ar + c * 32 + half * 8);
#pragma unroll
      for (int t = 0; t < 4; ++t) {
        v16bf b = frag_from_bf16(bbuf + (t * 16 + r) * ROWPAD + c * 32 + half * 8);
        acc[t] = WMMA_BF16(a, b, acc[t]);
      }
    }
    __syncthreads();
  }
  float* dst = part + (long)ks * BG * HH;
#pragma unroll
  for (int t = 0; t < 4; ++t)
    store_tile_f32(dst, HH, m0, n0 + t * 16, acc[t], lane);
}

__global__ void k_h0_fin(const float* __restrict__ part, const float* __restrict__ bias,
                         bf16* __restrict__ h0) {
  int t = blockIdx.x * blockDim.x + threadIdx.x;
  if (t < BG * HH) {
    float s = bias[t & (HH - 1)];
#pragma unroll
    for (int ks = 0; ks < 8; ++ks) s += part[(long)ks * BG * HH + t];
    h0[t] = (bf16)fmaxf(s, 0.0f);
  }
}

// ---------------------------------------------------------------------------
// Stage 4: h1[256][1024] = relu(h0 @ W_h1 + b_h1)
// ---------------------------------------------------------------------------
__global__ __launch_bounds__(128) void k_h1(const bf16* __restrict__ A,
                                            const bf16* __restrict__ W,
                                            const float* __restrict__ bias,
                                            bf16* __restrict__ h1) {
  int lane = threadIdx.x & 31;
  int gw = blockIdx.x * (blockDim.x >> 5) + (threadIdx.x >> 5);
  int mt = gw >> 4, ns = gw & 15;                    // 16 x 16
  int m0 = mt * 16, n0 = ns * 64;
  int half = lane >> 4, r = lane & 15;
  v8f acc[4] = {};
  for (int k0 = 0; k0 < HH; k0 += 32) {
    v16bf a = frag_from_bf16(A + (long)(m0 + r) * HH + k0 + half * 8);
#pragma unroll
    for (int t = 0; t < 4; ++t) {
      v16bf b = frag_from_bf16(W + (long)(n0 + t * 16 + r) * HH + k0 + half * 8);
      acc[t] = WMMA_BF16(a, b, acc[t]);
    }
  }
#pragma unroll
  for (int t = 0; t < 4; ++t)
    store_tile_bf16(h1, HH, m0, n0 + t * 16, acc[t], bias, true, lane);
}

// ---------------------------------------------------------------------------
// Stage 5: out[256] = h1 @ W_out + b_out   (wave-per-row dot, shfl reduce)
// ---------------------------------------------------------------------------
__global__ __launch_bounds__(128) void k_out(const bf16* __restrict__ h1,
                                             const float* __restrict__ Wo,
                                             const float* __restrict__ bo,
                                             float* __restrict__ out) {
  int lane = threadIdx.x & 31;
  int row = blockIdx.x * (blockDim.x >> 5) + (threadIdx.x >> 5);
  float s = 0.0f;
  for (int i = lane; i < HH; i += 32)
    s += (float)h1[(long)row * HH + i] * Wo[i];
#pragma unroll
  for (int off = 16; off; off >>= 1) s += __shfl_xor(s, off, 32);
  if (lane == 0) out[row] = s + bo[0];
}

// ---------------------------------------------------------------------------
extern "C" void kernel_launch(void* const* d_in, const int* in_sizes, int n_in,
                              void* d_out, int out_size, void* d_ws, size_t ws_size,
                              hipStream_t stream) {
  const float* node_attr = (const float*)d_in[0];
  const float* W_node    = (const float*)d_in[1];
  const float* b_node    = (const float*)d_in[2];
  const float* W_conv    = (const float*)d_in[3];
  const float* b_conv    = (const float*)d_in[4];
  const float* W_h0      = (const float*)d_in[5];
  const float* b_h0      = (const float*)d_in[6];
  const float* W_h1      = (const float*)d_in[7];
  const float* b_h1      = (const float*)d_in[8];
  const float* W_out     = (const float*)d_in[9];
  const float* b_out     = (const float*)d_in[10];
  const int*   frame_idx = (const int*)d_in[11];
  float* out = (float*)d_out;

  // Workspace carve-up (256B aligned)
  char* ws = (char*)d_ws;
  size_t off = 0;
  auto carve = [&](size_t bytes) -> char* {
    char* p = ws + off;
    off = (off + bytes + 255) & ~(size_t)255;
    return p;
  };
  bf16*  WnT  = (bf16*)carve((size_t)DD * DIN * sizeof(bf16));       //  64 KB
  bf16*  WcT  = (bf16*)carve((size_t)DD * KC * sizeof(bf16));        //   2 MB
  bf16*  Wh0T = (bf16*)carve((size_t)HH * KTOT * sizeof(bf16));      //  32 MB
  bf16*  Wh1T = (bf16*)carve((size_t)HH * HH * sizeof(bf16));        //   2 MB
  bf16*  x    = (bf16*)carve((size_t)NNODE * DD * sizeof(bf16));     //  16 MB
  bf16*  hc   = (bf16*)carve((size_t)FR * DD * sizeof(bf16));        //   8 MB
  float* part = (float*)carve((size_t)8 * BG * HH * sizeof(float));  //   8 MB
  bf16*  h0   = (bf16*)carve((size_t)BG * HH * sizeof(bf16));        // 512 KB
  bf16*  h1   = (bf16*)carve((size_t)BG * HH * sizeof(bf16));        // 512 KB
  (void)ws_size; (void)in_sizes; (void)n_in; (void)out_size;

  // Weight prep: f32 [K][N] -> bf16 [N][K]
  k_transpose_bf16<<<(DIN * DD + 255) / 256, 256, 0, stream>>>(W_node, WnT, DIN, DD);
  k_transpose_bf16<<<(KC * DD + 255) / 256, 256, 0, stream>>>(W_conv, WcT, KC, DD);
  k_transpose_bf16<<<(KTOT * HH + 255) / 256, 256, 0, stream>>>(W_h0, Wh0T, KTOT, HH);
  k_transpose_bf16<<<(HH * HH + 255) / 256, 256, 0, stream>>>(W_h1, Wh1T, HH, HH);

  // Stage 1: node embedding (8192 waves)
  k_embed<<<2048, 128, 0, stream>>>(node_attr, WnT, b_node, x);
  // Stage 2: patchy conv, TDM-staged B + fused gather (4096 waves)
  k_conv<<<1024, 128, 0, stream>>>(x, frame_idx, WcT, b_conv, hc);
  // Stage 3: h0 split-K, TDM-staged B (2048 waves) + finalize
  k_h0_part<<<512, 128, 0, stream>>>(hc, Wh0T, part);
  k_h0_fin<<<(BG * HH + 255) / 256, 256, 0, stream>>>(part, b_h0, h0);
  // Stage 4: h1 (256 waves)
  k_h1<<<64, 128, 0, stream>>>(h0, Wh1T, b_h1, h1);
  // Stage 5: final projection
  k_out<<<64, 128, 0, stream>>>(h1, W_out, b_out, out);
}
